// DPS_2808908612105
// MI455X (gfx1250) — matrix operand order repaired and checked by reading, not compiled, not run on hardware
//
#include <hip/hip_runtime.h>
#include <hip/hip_bf16.h>
#include <stdint.h>

typedef __attribute__((ext_vector_type(16))) _Float16 v16h;
typedef __attribute__((ext_vector_type(8)))  _Float16 v8h;
typedef __attribute__((ext_vector_type(8)))  float    v8f;

#define NB     8      // batch
#define SD     256    // flattened score dim (16x16)
#define NSAMP  500
#define TOPK   10
#define KPAD   16     // pad K=10 -> 16 for WMMA M dim
#define NC     3
#define IMG    1024
#define SCALE  64
#define PATCH  128
#define PADOFF 32     // (PATCH - SCALE)/2
#define SIGMA  0.05f

#define BROWS  16
#define BCOLS  1088   // source columns gc in [-32, 1056); col = gc + 32

// ---------------------------------------------------------------------------
// Kernel A: per-batch softmax entropy + min/max normalization of scores.
// ---------------------------------------------------------------------------
__global__ void kA_scores(const float* __restrict__ scores,
                          float* __restrict__ norm,
                          float* __restrict__ entrb) {
    __shared__ float sb[SD];
    const int b = blockIdx.x;
    const int t = threadIdx.x;
    const float s = scores[b * SD + t];

    sb[t] = s; __syncthreads();
    for (int o = 128; o > 0; o >>= 1) { if (t < o) sb[t] = fmaxf(sb[t], sb[t + o]); __syncthreads(); }
    const float smax = sb[0];
    __syncthreads();

    sb[t] = expf(s - smax); __syncthreads();
    for (int o = 128; o > 0; o >>= 1) { if (t < o) sb[t] += sb[t + o]; __syncthreads(); }
    const float L = smax + logf(sb[0]);   // logsumexp
    __syncthreads();

    sb[t] = expf(s - L) * s; __syncthreads();
    for (int o = 128; o > 0; o >>= 1) { if (t < o) sb[t] += sb[t + o]; __syncthreads(); }
    const float entr = L - sb[0];
    __syncthreads();

    sb[t] = s; __syncthreads();
    for (int o = 128; o > 0; o >>= 1) { if (t < o) sb[t] = fminf(sb[t], sb[t + o]); __syncthreads(); }
    const float smin = sb[0];

    norm[b * SD + t] = (s - smin) / (smax - smin + 1e-5f);
    if (t == 0) entrb[b] = entr;
}

// ---------------------------------------------------------------------------
// Kernel B: perturbed top-k. One wave32 per (batch, sample).
// ---------------------------------------------------------------------------
__device__ inline uint32_t pcg_hash(uint32_t x) {
    x = x * 747796405u + 2891336453u;
    uint32_t w = ((x >> ((x >> 28u) + 4u)) ^ x) * 277803737u;
    return (w >> 22u) ^ w;
}

__global__ void kB_topk(const float* __restrict__ norm, int* __restrict__ topidx) {
    const int wid  = blockIdx.x;          // 0 .. NB*NSAMP-1
    const int b    = wid / NSAMP;
    const int n    = wid % NSAMP;
    const int lane = threadIdx.x;         // wave32

    float vals[8];
#pragma unroll
    for (int e = 0; e < 8; ++e) {
        const int idx = e * 32 + lane;
        const uint32_t seed = (uint32_t)((b * NSAMP + n) * SD + idx);
        const uint32_t h1 = pcg_hash(seed * 0x9E3779B1u + 0x85EBCA77u);
        const uint32_t h2 = pcg_hash(h1 ^ 0x27D4EB2Fu);
        const float u1 = ((float)h1 + 1.0f) * 2.3283064e-10f;   // (0,1]
        const float u2 = ((float)h2) * 2.3283064e-10f;
        const float z  = sqrtf(-2.0f * logf(u1)) * __cosf(6.2831853f * u2);
        vals[e] = norm[b * SD + idx] + SIGMA * z;
    }

    int top[TOPK];
    for (int t = 0; t < TOPK; ++t) {
        float best = -1e30f; int bidx = SD;
#pragma unroll
        for (int e = 0; e < 8; ++e) {
            const int idx = e * 32 + lane;
            if (vals[e] > best || (vals[e] == best && idx < bidx)) { best = vals[e]; bidx = idx; }
        }
#pragma unroll
        for (int off = 16; off > 0; off >>= 1) {
            const float ov = __shfl_xor(best, off, 32);
            const int   oi = __shfl_xor(bidx, off, 32);
            if (ov > best || (ov == best && oi < bidx)) { best = ov; bidx = oi; }
        }
        top[t] = bidx;
#pragma unroll
        for (int e = 0; e < 8; ++e)
            if ((bidx >> 5) == e && (bidx & 31) == lane) vals[e] = -1e30f;
    }

    // sort indices ascending (compare-exchange, constant indexing only)
#pragma unroll
    for (int pass = 0; pass < TOPK; ++pass)
#pragma unroll
        for (int i = 0; i < TOPK - 1; ++i)
            if (top[i] > top[i + 1]) { const int tmp = top[i]; top[i] = top[i + 1]; top[i + 1] = tmp; }

#pragma unroll
    for (int k2 = 0; k2 < TOPK; ++k2)
        if (lane == k2) topidx[(b * NSAMP + n) * TOPK + k2] = top[k2];
}

// ---------------------------------------------------------------------------
// Kernel B2: deterministic histogram -> ind (f16, WMMA A operand precision)
// ---------------------------------------------------------------------------
__global__ void kB2_hist(const int* __restrict__ topidx, _Float16* __restrict__ indh) {
    const int gid = blockIdx.x * blockDim.x + threadIdx.x;   // NB*KPAD*SD
    const int ij = gid & (SD - 1);
    const int kp = (gid >> 8) & (KPAD - 1);
    const int b  = gid >> 12;
    float v = 0.0f;
    if (kp < TOPK) {
        int cnt = 0;
        const int* tp = topidx + b * NSAMP * TOPK + kp;
        for (int n = 0; n < NSAMP; ++n) cnt += (tp[n * TOPK] == ij);
        v = (float)cnt * (1.0f / NSAMP);
    }
    indh[gid] = (_Float16)v;
}

// ---------------------------------------------------------------------------
// Kernel C: patches[b,k,c,h,w] = sum_ij ind[b,k,ij] * xpad[b,c,i*64+h,j*64+w]
// Per block (b,c,h): fill LDS in direct-load B layout (contiguous k order) (scatter at fill time:
// each padded source element lands in its <=2 (j,w) window aliases), then
// 8 waves each run 8 chained v_wmma_f32_16x16x32_f16 where BOTH operands are
// two aligned 16B vector loads (A: global_load_b128, B: ds_load_b128).
//
// sBN layout: half jhi (j>=8) at +16384 halves; element (i, w, jlo) at
// (i*128 + w)*8 + jlo  -> per-lane 16B runs, dword stride 4 per w
// (conflict-free across lanes 0-15). Total LDS = 64 KB.
// ---------------------------------------------------------------------------
__global__ void kC_patches(const float* __restrict__ x,
                           const _Float16* __restrict__ indh,
                           float* __restrict__ out) {
    __shared__ __align__(16) _Float16 sBN[2 * BROWS * PATCH * 8]; // 65536 B

    const int blk = blockIdx.x;                 // NB*NC*PATCH
    const int h   = blk & (PATCH - 1);
    const int c   = (blk / PATCH) % NC;
    const int b   = blk / (PATCH * NC);
    const int t   = threadIdx.x;

    const float* __restrict__ xb = x + ((size_t)(b * NC + c)) * IMG * IMG;

    // ---- fill: coalesced read of padded rows, scatter to window aliases ----
    for (int e = t; e < BROWS * BCOLS; e += 256) {
        const int row = e / BCOLS;              // i = 0..15
        const int col = e - row * BCOLS;        // gc + 32, 0..1087
        const int gr = row * SCALE + h - PADOFF;
        const int gc = col - PADOFF;
        float v = 0.0f;
        if ((unsigned)gr < (unsigned)IMG && (unsigned)gc < (unsigned)IMG)
            v = xb[gr * IMG + gc];
        const _Float16 hv = (_Float16)v;

        const int j1 = col >> 6;                // alias 1: w in [0,64)
        const int w1 = col & 63;
        if (j1 < 16)
            sBN[((j1 & 8) << 11) + (row * PATCH + w1) * 8 + (j1 & 7)] = hv;
        const int j2 = j1 - 1;                  // alias 2: w in [64,128)
        if (j2 >= 0)
            sBN[((j2 & 8) << 11) + (row * PATCH + w1 + 64) * 8 + (j2 & 7)] = hv;
    }
    __syncthreads();

    const int lane = t & 31;
    const int wave = t >> 5;                    // 8 waves = 8 w-tiles
    const int w    = wave * 16 + (lane & 15);
    const int mlow = lane & 15;
    const int aoff = (lane & 16) ? 16 : 0;      // A: K offset for upper lanes
    const int bsel = (lane & 16) ? 1 : 0;       // B: upper lanes take K+16 row

    const _Float16* __restrict__ ag = indh + b * KPAD * SD + mlow * SD;

    v8f acc = {0.f, 0.f, 0.f, 0.f, 0.f, 0.f, 0.f, 0.f};

#pragma unroll
    for (int kc = 0; kc < 8; ++kc) {
        // A operand: halves 0-15 = K[base..base+15]
        const v8h alo = *(const v8h*)(ag + kc * 32 + aoff);
        const v8h ahi = *(const v8h*)(ag + kc * 32 + aoff + 8);
        const v16h a = __builtin_shufflevector(alo, ahi,
            0, 1, 2, 3, 4, 5, 6, 7, 8, 9, 10, 11, 12, 13, 14, 15);

        // B operand: image row i = 2*kc + bsel; 16 j-values = two 16B LDS runs
        const _Float16* bp = &sBN[((2 * kc + bsel) * PATCH + w) * 8];
        const v8h blo = *(const v8h*)(bp);            // j = 0..7
        const v8h bhi = *(const v8h*)(bp + 16384);    // j = 8..15
        const v16h bm = __builtin_shufflevector(blo, bhi,
            0, 1, 2, 3, 4, 5, 6, 7, 8, 9, 10, 11, 12, 13, 14, 15);

        acc = __builtin_amdgcn_wmma_f32_16x16x32_f16(
            /*neg_a=*/false, a, /*neg_b=*/false, bm,
            /*c_mod=*/(short)0, acc, /*reuse_a=*/false, /*reuse_b=*/false);
    }

    // D layout: VGPR r -> M = r (+8 for lanes 16..31), N = lane&15
    const int mbase = (lane & 16) ? 8 : 0;
#pragma unroll
    for (int r = 0; r < 8; ++r) {
        const int mk = mbase + r;
        if (mk < TOPK)
            out[((((size_t)b * TOPK + mk) * NC + c) * PATCH + h) * PATCH + w] = acc[r];
    }
}

// ---------------------------------------------------------------------------
// Kernel D: mean of per-batch entropies -> scalar output
// ---------------------------------------------------------------------------
__global__ void kD_entr(const float* __restrict__ entrb, float* __restrict__ out_entr) {
    const int lane = threadIdx.x;
    float v = (lane < NB) ? entrb[lane] : 0.0f;
#pragma unroll
    for (int off = 16; off > 0; off >>= 1) v += __shfl_xor(v, off, 32);
    if (lane == 0) *out_entr = v * (1.0f / NB);
}

// ---------------------------------------------------------------------------
extern "C" void kernel_launch(void* const* d_in, const int* in_sizes, int n_in,
                              void* d_out, int out_size, void* d_ws, size_t ws_size,
                              hipStream_t stream) {
    (void)in_sizes; (void)n_in; (void)ws_size;
    const float* x      = (const float*)d_in[0];   // (8,3,1024,1024)
    const float* scores = (const float*)d_in[1];   // (8,16,16)

    float*     ws     = (float*)d_ws;
    float*     norm   = ws;                                  // 2048 f
    float*     entrb  = ws + 2048;                           // 8 f
    int*       topidx = (int*)(ws + 2112);                   // 40000 i
    _Float16*  indh   = (_Float16*)(ws + 2112 + NB * NSAMP * TOPK); // 32768 h, 16B aligned

    float* patches = (float*)d_out;
    float* entr    = patches + (out_size - 1);

    kA_scores <<<NB, SD, 0, stream>>>(scores, norm, entrb);
    kB_topk   <<<NB * NSAMP, 32, 0, stream>>>(norm, topidx);
    kB2_hist  <<<(NB * KPAD * SD) / 256, 256, 0, stream>>>(topidx, indh);
    kC_patches<<<NB * NC * PATCH, 256, 0, stream>>>(x, indh, patches);
    kD_entr   <<<1, 32, 0, stream>>>(entrb, entr);
}
